// Net_16673063043119
// MI455X (gfx1250) — matrix operand order repaired and checked by reading, not compiled, not run on hardware
//
#include <hip/hip_runtime.h>
#include <hip/hip_bf16.h>

// ---------------------------------------------------------------------------
// GraphSAGE 2-layer forward for MI455X (gfx1250, wave32, WMMA).
// GEMMs run on v_wmma_f32_16x16x32_bf16; aggregation via L2-resident f32
// atomics (x / agg buffers fit in the 192MB L2, so scatter never hits HBM).
// ---------------------------------------------------------------------------

typedef __attribute__((ext_vector_type(16))) __bf16 v16bf;
typedef __attribute__((ext_vector_type(8)))  float  v8f;

#define N_NODES 50000
#define N_EDGES 800000
#define F_INF   128
#define HIDF    128
#define N_CLS   40
#define M_TILES (N_NODES / 16)   // 3125 exact

union ABFrag { uint4 u[2]; v16bf v; };

static __device__ __forceinline__ unsigned short f32_to_bf16(float f) {
    union { float f; unsigned u; } v; v.f = f;
    unsigned u = v.u;
    u += 0x7FFFu + ((u >> 16) & 1u);      // round-to-nearest-even
    return (unsigned short)(u >> 16);
}

static __device__ __forceinline__ unsigned pack_bf16x2(float lo, float hi) {
    return (unsigned)f32_to_bf16(lo) | ((unsigned)f32_to_bf16(hi) << 16);
}

// ---------------------------------------------------------------- zero fill
__global__ void zero_f32(float* __restrict__ p, int n) {
    int i = blockIdx.x * blockDim.x + threadIdx.x;
    if (i < n) p[i] = 0.0f;
}

// -------------------------------------------------- mean-aggregate scatter
// one thread per (edge, feature-pair): 64 threads cover the 128 features.
__global__ void scatter_feats(const float* __restrict__ feat,
                              const int* __restrict__ src,
                              const int* __restrict__ dst,
                              float* __restrict__ agg,
                              float* __restrict__ deg,
                              int nEdges, int doDeg) {
    long long gid = (long long)blockIdx.x * blockDim.x + threadIdx.x;
    int e  = (int)(gid >> 6);
    int fp = ((int)gid & 63) * 2;
    if (e >= nEdges) return;
    int s = src[e], d = dst[e];
    float2 v = *(const float2*)(feat + (size_t)s * F_INF + fp);
    atomicAdd(&agg[(size_t)d * F_INF + fp],     v.x);
    atomicAdd(&agg[(size_t)d * F_INF + fp + 1], v.y);
    if (doDeg && fp == 0) atomicAdd(&deg[d], 1.0f);
}

// -------------------------------------------------------- f32 -> bf16 copy
__global__ void cvt_bf16(const float* __restrict__ in,
                         unsigned short* __restrict__ out, int n) {
    int i = blockIdx.x * blockDim.x + threadIdx.x;
    if (i < n) out[i] = f32_to_bf16(in[i]);
}

// ----------------------------------- agg / max(deg,1) -> bf16 (normalized)
__global__ void norm_cvt_bf16(const float* __restrict__ agg,
                              const float* __restrict__ deg,
                              unsigned short* __restrict__ out, int n) {
    int i = blockIdx.x * blockDim.x + threadIdx.x;
    if (i >= n) return;
    int node = i >> 7;                    // /128
    float d = fmaxf(deg[node], 1.0f);
    out[i] = f32_to_bf16(agg[i] / d);
}

// ------------------------------------------------ weight pack to B-fragment
// B-matrix 32x16 bf16 WMMA layout: lane = {khalf[1], n[3:0]}, VGPR j holds
// packed K pair (ksel + 2j + (j>=4 ? 8 : 0), +1) with ksel = 8*khalf, for column n.  One dword per thread.
__global__ void pack_weight(const float* __restrict__ W,
                            unsigned* __restrict__ P,
                            int nTot, int nTiles) {
    int idx = blockIdx.x * blockDim.x + threadIdx.x;
    int total = 4 * nTiles * 32 * 8;      // ktiles * ntiles * lanes * dwords
    if (idx >= total) return;
    int j    = idx & 7;
    int lane = (idx >> 3) & 31;
    int t    = idx >> 8;
    int nt   = t % nTiles;
    int kt   = t / nTiles;
    int n    = lane & 15;
    int ksel = (lane >> 4) * 8;
    int col  = nt * 16 + n;
    int k0   = kt * 32 + ksel + 2 * j + (j >= 4 ? 8 : 0);
    float f0 = 0.0f, f1 = 0.0f;
    if (col < nTot) {
        f0 = W[(size_t)k0 * nTot + col];
        f1 = W[(size_t)(k0 + 1) * nTot + col];
    }
    P[idx] = pack_bf16x2(f0, f1);
}

// ---------------------------------------------------------- layer-1 GEMM
// h = relu(aggn @ W_l1 + x @ W_r1 + b1); writes f32 (for next scatter) and
// bf16 (for next GEMM).  8 waves/block = 8 N-tiles, grid = 3125 M-tiles.
__global__ void __launch_bounds__(256)
sage1_gemm(const unsigned short* __restrict__ xb,
           const unsigned short* __restrict__ ab,
           const unsigned* __restrict__ Wl,
           const unsigned* __restrict__ Wr,
           const float* __restrict__ bias,
           float* __restrict__ h32,
           unsigned short* __restrict__ h16) {
    const int wave  = threadIdx.x >> 5;      // 0..7 = N-tile
    const int lane  = threadIdx.x & 31;
    const int mtile = blockIdx.x;
    const int m     = lane & 15;
    const int half  = lane >> 4;
    const int khalf = half * 8;
    const int rowA  = mtile * 16 + m;

    v8f c;
    float bv = bias[wave * 16 + m];          // col == lane&15
    #pragma unroll
    for (int r = 0; r < 8; ++r) c[r] = bv;

    #pragma unroll
    for (int kt = 0; kt < 4; ++kt) {
        const int K0 = kt * 32;
        ABFrag a1, a2, wb1, wb2;
        const unsigned short* ar = ab + (size_t)rowA * HIDF + K0 + khalf;
        a1.u[0] = *(const uint4*)(ar);
        a1.u[1] = *(const uint4*)(ar + 16);
        const unsigned short* xr = xb + (size_t)rowA * HIDF + K0 + khalf;
        a2.u[0] = *(const uint4*)(xr);
        a2.u[1] = *(const uint4*)(xr + 16);
        const unsigned* wl = Wl + ((size_t)(kt * 8 + wave) * 32 + lane) * 8;
        wb1.u[0] = *(const uint4*)(wl);
        wb1.u[1] = *(const uint4*)(wl + 4);
        const unsigned* wr = Wr + ((size_t)(kt * 8 + wave) * 32 + lane) * 8;
        wb2.u[0] = *(const uint4*)(wr);
        wb2.u[1] = *(const uint4*)(wr + 4);
        c = __builtin_amdgcn_wmma_f32_16x16x32_bf16(false, a1.v, false, wb1.v,
                                                    (short)0, c, false, false);
        c = __builtin_amdgcn_wmma_f32_16x16x32_bf16(false, a2.v, false, wb2.v,
                                                    (short)0, c, false, false);
    }

    #pragma unroll
    for (int r = 0; r < 8; ++r) {
        float v  = fmaxf(c[r], 0.0f);        // ReLU
        int row  = mtile * 16 + r + half * 8;
        int col  = wave * 16 + m;
        h32[(size_t)row * HIDF + col] = v;
        h16[(size_t)row * HIDF + col] = f32_to_bf16(v);
    }
}

// ---------------------------------------------------------- layer-2 GEMM
// logits = aggn2 @ W_l2 + h @ W_r2 + b2, N padded 40->48 (3 waves/block).
__global__ void __launch_bounds__(96)
sage2_gemm(const unsigned short* __restrict__ hb,
           const unsigned short* __restrict__ ab,
           const unsigned* __restrict__ Wl,
           const unsigned* __restrict__ Wr,
           const float* __restrict__ bias,
           float* __restrict__ logits) {
    const int wave  = threadIdx.x >> 5;      // 0..2 = N-tile
    const int lane  = threadIdx.x & 31;
    const int mtile = blockIdx.x;
    const int m     = lane & 15;
    const int half  = lane >> 4;
    const int khalf = half * 8;
    const int rowA  = mtile * 16 + m;
    const int gcol  = wave * 16 + m;

    v8f c;
    float bv = (gcol < N_CLS) ? bias[gcol] : 0.0f;
    #pragma unroll
    for (int r = 0; r < 8; ++r) c[r] = bv;

    #pragma unroll
    for (int kt = 0; kt < 4; ++kt) {
        const int K0 = kt * 32;
        ABFrag a1, a2, wb1, wb2;
        const unsigned short* ar = ab + (size_t)rowA * HIDF + K0 + khalf;
        a1.u[0] = *(const uint4*)(ar);
        a1.u[1] = *(const uint4*)(ar + 16);
        const unsigned short* hr = hb + (size_t)rowA * HIDF + K0 + khalf;
        a2.u[0] = *(const uint4*)(hr);
        a2.u[1] = *(const uint4*)(hr + 16);
        const unsigned* wl = Wl + ((size_t)(kt * 3 + wave) * 32 + lane) * 8;
        wb1.u[0] = *(const uint4*)(wl);
        wb1.u[1] = *(const uint4*)(wl + 4);
        const unsigned* wr = Wr + ((size_t)(kt * 3 + wave) * 32 + lane) * 8;
        wb2.u[0] = *(const uint4*)(wr);
        wb2.u[1] = *(const uint4*)(wr + 4);
        c = __builtin_amdgcn_wmma_f32_16x16x32_bf16(false, a1.v, false, wb1.v,
                                                    (short)0, c, false, false);
        c = __builtin_amdgcn_wmma_f32_16x16x32_bf16(false, a2.v, false, wb2.v,
                                                    (short)0, c, false, false);
    }

    if (gcol < N_CLS) {
        #pragma unroll
        for (int r = 0; r < 8; ++r) {
            int row = mtile * 16 + r + half * 8;
            logits[(size_t)row * N_CLS + gcol] = c[r];
        }
    }
}

// ------------------------------------------------------------ log-softmax
__global__ void logsoftmax_k(const float* __restrict__ logits,
                             float* __restrict__ out, int n) {
    int i = blockIdx.x * blockDim.x + threadIdx.x;
    if (i >= n) return;
    const float* r = logits + (size_t)i * N_CLS;
    float mx = -3.4e38f;
    #pragma unroll
    for (int j = 0; j < N_CLS; ++j) mx = fmaxf(mx, r[j]);
    float s = 0.0f;
    #pragma unroll
    for (int j = 0; j < N_CLS; ++j) s += __expf(r[j] - mx);
    float lse = __logf(s) + mx;
    #pragma unroll
    for (int j = 0; j < N_CLS; ++j)
        out[(size_t)i * N_CLS + j] = r[j] - lse;
}

// ---------------------------------------------------------------------------
extern "C" void kernel_launch(void* const* d_in, const int* in_sizes, int n_in,
                              void* d_out, int out_size, void* d_ws, size_t ws_size,
                              hipStream_t stream) {
    (void)in_sizes; (void)n_in; (void)out_size; (void)ws_size;
    const float* x   = (const float*)d_in[0];
    const int*   ei  = (const int*)d_in[1];
    const float* Wl1 = (const float*)d_in[2];
    const float* b1  = (const float*)d_in[3];
    const float* Wr1 = (const float*)d_in[4];
    const float* Wl2 = (const float*)d_in[5];
    const float* b2  = (const float*)d_in[6];
    const float* Wr2 = (const float*)d_in[7];
    const int* src = ei;
    const int* dst = ei + N_EDGES;

    // workspace carve-up (256B aligned)
    char* ws = (char*)d_ws;
    size_t off = 0;
    auto carve = [&](size_t bytes) {
        char* p = ws + off;
        off = (off + bytes + 255) & ~(size_t)255;
        return p;
    };
    const size_t NF = (size_t)N_NODES * F_INF;
    float*          deg    = (float*)carve(N_NODES * sizeof(float));
    float*          agg    = (float*)carve(NF * sizeof(float));          // reused for both layers
    float*          h32    = (float*)carve(NF * sizeof(float));
    unsigned short* xb     = (unsigned short*)carve(NF * 2);
    unsigned short* aggnb  = (unsigned short*)carve(NF * 2);             // reused
    unsigned short* hb     = (unsigned short*)carve(NF * 2);
    unsigned*       Wl1p   = (unsigned*)carve(4 * 8 * 32 * 8 * 4);
    unsigned*       Wr1p   = (unsigned*)carve(4 * 8 * 32 * 8 * 4);
    unsigned*       Wl2p   = (unsigned*)carve(4 * 3 * 32 * 8 * 4);
    unsigned*       Wr2p   = (unsigned*)carve(4 * 3 * 32 * 8 * 4);
    float*          logits = (float*)carve((size_t)N_NODES * N_CLS * sizeof(float));
    float*          out    = (float*)d_out;

    const int TB = 256;
    dim3 blk(TB);

    // --- init + weight packing + x conversion -----------------------------
    zero_f32<<<(N_NODES + TB - 1) / TB, blk, 0, stream>>>(deg, N_NODES);
    zero_f32<<<(int)((NF + TB - 1) / TB), blk, 0, stream>>>(agg, (int)NF);
    pack_weight<<<(4 * 8 * 32 * 8 + TB - 1) / TB, blk, 0, stream>>>(Wl1, Wl1p, HIDF, 8);
    pack_weight<<<(4 * 8 * 32 * 8 + TB - 1) / TB, blk, 0, stream>>>(Wr1, Wr1p, HIDF, 8);
    pack_weight<<<(4 * 3 * 32 * 8 + TB - 1) / TB, blk, 0, stream>>>(Wl2, Wl2p, N_CLS, 3);
    pack_weight<<<(4 * 3 * 32 * 8 + TB - 1) / TB, blk, 0, stream>>>(Wr2, Wr2p, N_CLS, 3);
    cvt_bf16<<<(int)((NF + TB - 1) / TB), blk, 0, stream>>>(x, xb, (int)NF);

    // --- layer 1 ----------------------------------------------------------
    long long sthreads = (long long)N_EDGES * 64;
    int sblocks = (int)((sthreads + TB - 1) / TB);
    scatter_feats<<<sblocks, blk, 0, stream>>>(x, src, dst, agg, deg, N_EDGES, 1);
    norm_cvt_bf16<<<(int)((NF + TB - 1) / TB), blk, 0, stream>>>(agg, deg, aggnb, (int)NF);
    sage1_gemm<<<M_TILES, 256, 0, stream>>>(xb, aggnb, Wl1p, Wr1p, b1, h32, hb);

    // --- layer 2 ----------------------------------------------------------
    zero_f32<<<(int)((NF + TB - 1) / TB), blk, 0, stream>>>(agg, (int)NF);
    scatter_feats<<<sblocks, blk, 0, stream>>>(h32, src, dst, agg, deg, N_EDGES, 0);
    norm_cvt_bf16<<<(int)((NF + TB - 1) / TB), blk, 0, stream>>>(agg, deg, aggnb, (int)NF);
    sage2_gemm<<<M_TILES, 96, 0, stream>>>(hb, aggnb, Wl2p, Wr2p, b2, logits);

    // --- log-softmax ------------------------------------------------------
    logsoftmax_k<<<(N_NODES + TB - 1) / TB, blk, 0, stream>>>(logits, out, N_NODES);
}
